// GCN_1872605741597
// MI455X (gfx1250) — matrix operand order repaired and checked by reading, not compiled, not run on hardware
//
#include <hip/hip_runtime.h>
#include <hip/hip_bf16.h>

// ---------------------------------------------------------------------------
// Types for CDNA5 WMMA
// ---------------------------------------------------------------------------
typedef __attribute__((ext_vector_type(16))) __bf16 v16bf;
typedef __attribute__((ext_vector_type(8)))  float  v8f;

union Frag {
    v16bf v;
    unsigned int u[8];
};

// Round-to-nearest-even fp32 -> bf16 (returned in low 16 bits)
__device__ __forceinline__ unsigned int f32_to_bf16_rne(float f) {
    unsigned int u = __float_as_uint(f);
    return (u + 0x7FFFu + ((u >> 16) & 1u)) >> 16;
}

// ---------------------------------------------------------------------------
// Weight pre-pack: W[K,128] fp32 -> bf16 hi/lo packed in B-fragment layout.
// B fragment as packed here (32x16, bf16): lanes 0-15 hold column n=lane, K=0..15
// (dword d = K pair {2d, 2d+1}); lanes 16-31 hold K=16..31.
// Buffer layout: [kchunk][ntile][lane][dword0..7]  (each lane: 32 contiguous B)
// ---------------------------------------------------------------------------
__global__ void pack_w_kernel(const float* __restrict__ W,
                              unsigned int* __restrict__ Whi,
                              unsigned int* __restrict__ Wlo, int K) {
    int tid = blockIdx.x * blockDim.x + threadIdx.x;
    int total = (K >> 1) * 128;
    if (tid >= total) return;
    int p = tid >> 7;      // k-pair index
    int n = tid & 127;
    int k = p << 1;

    float w0 = W[(size_t)k * 128 + n];
    float w1 = W[(size_t)(k + 1) * 128 + n];

    int kc   = k >> 5;
    int kin  = k & 31;
    int half = kin >> 4;          // which 16-lane group
    int d    = (kin & 15) >> 1;   // dword within fragment
    int t    = n >> 4;            // 16-wide N tile
    int lane = half * 16 + (n & 15);
    size_t idx = ((((size_t)kc * 8 + t) * 32) + lane) * 8 + d;

    unsigned int h0 = f32_to_bf16_rne(w0);
    unsigned int h1 = f32_to_bf16_rne(w1);
    Whi[idx] = (h1 << 16) | (h0 & 0xFFFFu);

    float r0 = w0 - __uint_as_float(h0 << 16);
    float r1 = w1 - __uint_as_float(h1 << 16);
    unsigned int l0 = f32_to_bf16_rne(r0);
    unsigned int l1 = f32_to_bf16_rne(r1);
    Wlo[idx] = (l1 << 16) | (l0 & 0xFFFFu);
}

// ---------------------------------------------------------------------------
// WMMA GEMM: Y[M,128] = X[M,K] @ W (K multiple of 32, M multiple of 16).
// Block = 128 threads = 4 waves; each wave computes a 16x128 strip.
// Split-precision: X and W as bf16 hi+lo; 3 WMMA products per (kchunk,ntile).
// ---------------------------------------------------------------------------
__global__ __launch_bounds__(128)
void gemm_bf16x3_kernel(const float* __restrict__ X,
                        const unsigned int* __restrict__ Wp_hi,
                        const unsigned int* __restrict__ Wp_lo,
                        float* __restrict__ Y, int M, int K) {
    const int lane    = threadIdx.x & 31;
    const int wave    = threadIdx.x >> 5;
    const int rowBase = (blockIdx.x * 4 + wave) * 16;
    if (rowBase >= M) return;                    // wave-uniform: EXEC stays all-1s
    const int halfK = lane >> 4;                 // 0 or 1
    const int m     = rowBase + (lane & 15);
    const float* xrow = X + (size_t)m * K;

    v8f acc[8];
#pragma unroll
    for (int t = 0; t < 8; ++t) acc[t] = (v8f){0.f,0.f,0.f,0.f,0.f,0.f,0.f,0.f};

    const int nkc = K >> 5;
    for (int kc = 0; kc < nkc; ++kc) {
        // --- A fragment: per lane 16 contiguous K in [16h, 16h+15]
        //     (matches packed B layout)
        float a[16];
        const float* p0 = xrow + kc * 32 + 16 * halfK;
#pragma unroll
        for (int j = 0; j < 8; ++j) a[j] = p0[j];
#pragma unroll
        for (int j = 0; j < 8; ++j) a[8 + j] = p0[8 + j];

        Frag Ahi, Alo;
#pragma unroll
        for (int d = 0; d < 8; ++d) {
            float f0 = a[2 * d], f1 = a[2 * d + 1];
            unsigned int h0 = f32_to_bf16_rne(f0);
            unsigned int h1 = f32_to_bf16_rne(f1);
            Ahi.u[d] = (h1 << 16) | (h0 & 0xFFFFu);
            float r0 = f0 - __uint_as_float(h0 << 16);
            float r1 = f1 - __uint_as_float(h1 << 16);
            unsigned int l0 = f32_to_bf16_rne(r0);
            unsigned int l1 = f32_to_bf16_rne(r1);
            Alo.u[d] = (l1 << 16) | (l0 & 0xFFFFu);
        }

#pragma unroll
        for (int t = 0; t < 8; ++t) {
            Frag Bhi, Blo;
            const unsigned int* bh = Wp_hi + ((((size_t)kc * 8 + t) * 32) + lane) * 8;
            const unsigned int* bl = Wp_lo + ((((size_t)kc * 8 + t) * 32) + lane) * 8;
#pragma unroll
            for (int d = 0; d < 8; ++d) { Bhi.u[d] = bh[d]; Blo.u[d] = bl[d]; }

            acc[t] = __builtin_amdgcn_wmma_f32_16x16x32_bf16(
                false, Ahi.v, false, Bhi.v, (short)0, acc[t], false, false);
            acc[t] = __builtin_amdgcn_wmma_f32_16x16x32_bf16(
                false, Ahi.v, false, Blo.v, (short)0, acc[t], false, false);
            acc[t] = __builtin_amdgcn_wmma_f32_16x16x32_bf16(
                false, Alo.v, false, Bhi.v, (short)0, acc[t], false, false);
        }
    }

    // C/D layout: VGPR r -> M=r (lanes 0-15) / M=r+8 (lanes 16-31), N=lane&15
    float* yb = Y + (size_t)rowBase * 128;
#pragma unroll
    for (int t = 0; t < 8; ++t) {
        int n = t * 16 + (lane & 15);
#pragma unroll
        for (int r = 0; r < 8; ++r) {
            int mrow = r + 8 * halfK;
            yb[(size_t)mrow * 128 + n] = acc[t][r];
        }
    }
}

// ---------------------------------------------------------------------------
// Support kernels
// ---------------------------------------------------------------------------
__global__ void fill_kernel(float* __restrict__ p, float v, int n) {
    int tid = blockIdx.x * blockDim.x + threadIdx.x;
    if (tid < n) p[tid] = v;
}

__global__ void deg_kernel(const int* __restrict__ dst, float* __restrict__ deg, int e) {
    int tid = blockIdx.x * blockDim.x + threadIdx.x;
    if (tid < e) atomicAdd(&deg[dst[tid]], 1.0f);
}

__global__ void rsqrt_kernel(float* __restrict__ d, int n) {
    int tid = blockIdx.x * blockDim.x + threadIdx.x;
    if (tid < n) d[tid] = rsqrtf(d[tid]);
}

// One edge per 32-lane group; each lane handles 4 channels (float4).
__global__ __launch_bounds__(256)
void scatter_kernel(const float* __restrict__ xl, const int* __restrict__ src,
                    const int* __restrict__ dst, const float* __restrict__ dinv,
                    float* __restrict__ agg, int nEdges) {
    long long tid = (long long)blockIdx.x * blockDim.x + threadIdx.x;
    int e    = (int)(tid >> 5);
    int lane = (int)(tid & 31);
    if (e >= nEdges) return;
    int s = src[e], d = dst[e];
    float nrm = dinv[s] * dinv[d];
    const float4* xp = (const float4*)(xl + (size_t)s * 128);
    float4 v = xp[lane];
    float* ap = agg + (size_t)d * 128 + lane * 4;
    atomicAdd(ap + 0, nrm * v.x);
    atomicAdd(ap + 1, nrm * v.y);
    atomicAdd(ap + 2, nrm * v.z);
    atomicAdd(ap + 3, nrm * v.w);
}

// h = relu(agg + dinv^2 * xl + b)
__global__ void combine_kernel(const float* __restrict__ agg, const float* __restrict__ xl,
                               const float* __restrict__ dinv, const float* __restrict__ bias,
                               float* __restrict__ out, int n) {
    int tid = blockIdx.x * blockDim.x + threadIdx.x;
    if (tid >= n * 128) return;
    int i = tid >> 7, c = tid & 127;
    float di = dinv[i];
    float v = agg[tid] + di * di * xl[tid] + bias[c];
    out[tid] = v > 0.f ? v : 0.f;
}

__global__ void pool_kernel(const float* __restrict__ h, const int* __restrict__ batch,
                            float* __restrict__ sums, int n) {
    int tid = blockIdx.x * blockDim.x + threadIdx.x;
    if (tid >= n * 128) return;
    int i = tid >> 7, c = tid & 127;
    atomicAdd(&sums[batch[i] * 128 + c], h[tid]);
}

__global__ void count_kernel(const int* __restrict__ batch, float* __restrict__ cnt, int n) {
    int tid = blockIdx.x * blockDim.x + threadIdx.x;
    if (tid < n) atomicAdd(&cnt[batch[tid]], 1.0f);
}

__global__ void final_kernel(const float* __restrict__ sums, const float* __restrict__ cnt,
                             const float* __restrict__ Wl, const float* __restrict__ bl,
                             float* __restrict__ out, int nGraphs) {
    int g = blockIdx.x * blockDim.x + threadIdx.x;
    if (g >= nGraphs) return;
    float c = cnt[g];
    c = c > 1.f ? c : 1.f;
    float acc = 0.f;
    for (int ch = 0; ch < 128; ++ch) acc += (sums[g * 128 + ch] / c) * Wl[ch];
    out[g] = acc + bl[0];
}

// ---------------------------------------------------------------------------
// Host launch
// ---------------------------------------------------------------------------
extern "C" void kernel_launch(void* const* d_in, const int* in_sizes, int n_in,
                              void* d_out, int out_size, void* d_ws, size_t ws_size,
                              hipStream_t stream) {
    const float* x     = (const float*)d_in[0];
    const int*   ei    = (const int*)d_in[1];
    const int*   batch = (const int*)d_in[2];
    const float* W1    = (const float*)d_in[3];
    const float* b1    = (const float*)d_in[4];
    const float* W2    = (const float*)d_in[5];
    const float* b2    = (const float*)d_in[6];
    const float* Wl    = (const float*)d_in[7];
    const float* bl    = (const float*)d_in[8];
    float* out = (float*)d_out;

    const int IN_C = 768, HID = 128;
    const int NN = in_sizes[0] / IN_C;
    const int NE = in_sizes[1] / 2;
    const int NG = out_size;

    const int* src = ei;
    const int* dst = ei + NE;

    float* ws = (float*)d_ws;
    size_t o = 0;
    float* B1   = ws + o; o += (size_t)NN * HID;     // xl / h1
    float* B2   = ws + o; o += (size_t)NN * HID;     // xl2 / h2
    float* B3   = ws + o; o += (size_t)NN * HID;     // agg scratch
    float* dinv = ws + o; o += NN;                   // deg -> dinv in place
    unsigned int* wp1h = (unsigned int*)(ws + o); o += (size_t)(IN_C / 2) * HID;
    unsigned int* wp1l = (unsigned int*)(ws + o); o += (size_t)(IN_C / 2) * HID;
    unsigned int* wp2h = (unsigned int*)(ws + o); o += (size_t)(HID / 2) * HID;
    unsigned int* wp2l = (unsigned int*)(ws + o); o += (size_t)(HID / 2) * HID;
    float* sums = ws + o; o += (size_t)NG * HID;
    float* cnt  = ws + o; o += NG;                   // contiguous with sums

    const int nc = NN * HID;

    // Pack weights to WMMA B-fragment layout (bf16 hi/lo)
    int tot1 = (IN_C / 2) * HID;
    pack_w_kernel<<<(tot1 + 255) / 256, 256, 0, stream>>>(W1, wp1h, wp1l, IN_C);
    int tot2 = (HID / 2) * HID;
    pack_w_kernel<<<(tot2 + 255) / 256, 256, 0, stream>>>(W2, wp2h, wp2l, HID);

    // Degree (with self loop) -> dinv
    fill_kernel<<<(NN + 255) / 256, 256, 0, stream>>>(dinv, 1.0f, NN);
    deg_kernel<<<(NE + 255) / 256, 256, 0, stream>>>(dst, dinv, NE);
    rsqrt_kernel<<<(NN + 255) / 256, 256, 0, stream>>>(dinv, NN);

    const int nStrips = (NN + 15) / 16;
    const int gblocks = (nStrips + 3) / 4;
    const int sblocks = (int)(((long long)NE * 32 + 255) / 256);

    // Layer 1
    gemm_bf16x3_kernel<<<gblocks, 128, 0, stream>>>(x, wp1h, wp1l, B1, NN, IN_C);
    fill_kernel<<<(nc + 255) / 256, 256, 0, stream>>>(B3, 0.f, nc);
    scatter_kernel<<<sblocks, 256, 0, stream>>>(B1, src, dst, dinv, B3, NE);
    combine_kernel<<<(nc + 255) / 256, 256, 0, stream>>>(B3, B1, dinv, b1, B1, NN);

    // Layer 2
    gemm_bf16x3_kernel<<<gblocks, 128, 0, stream>>>(B1, wp2h, wp2l, B2, NN, HID);
    fill_kernel<<<(nc + 255) / 256, 256, 0, stream>>>(B3, 0.f, nc);
    scatter_kernel<<<sblocks, 256, 0, stream>>>(B2, src, dst, dinv, B3, NE);
    combine_kernel<<<(nc + 255) / 256, 256, 0, stream>>>(B3, B2, dinv, b2, B2, NN);

    // Global mean pool + final linear
    int np = NG * HID + NG;
    fill_kernel<<<(np + 255) / 256, 256, 0, stream>>>(sums, 0.f, np);
    pool_kernel<<<(nc + 255) / 256, 256, 0, stream>>>(B2, batch, sums, NN);
    count_kernel<<<(NN + 255) / 256, 256, 0, stream>>>(batch, cnt, NN);
    final_kernel<<<(NG + 63) / 64, 64, 0, stream>>>(sums, cnt, Wl, bl, out, NG);
}
